// DiffBeamTreeCell_transparent_31258771981076
// MI455X (gfx1250) — matrix operand order repaired and checked by reading, not compiled, not run on hardware
//
#include <hip/hip_runtime.h>
#include <hip/hip_bf16.h>
#include <math.h>

typedef _Float16 h16;
typedef __attribute__((ext_vector_type(16))) _Float16 v16h;
typedef __attribute__((ext_vector_type(8)))  _Float16 v8h;
typedef __attribute__((ext_vector_type(8)))  float    v8f;

#define NB 16
#define S0 512
#define SS 511
#define MROW2 (NB * SS)   // 8176

// ---------------------------------------------------------------------------
// WMMA operand loaders
// A (16x32 f16, row-major source): lanes 0-15 row=lane, K = {0..7,16..23};
// lanes 16-31 same row (lane&15), K = {8..15,24..31}.
__device__ __forceinline__ v16h load_a(const h16* rowp, int kbase, int lane) {
  int base = kbase + ((lane & 16) ? 8 : 0);
  v8h lo = *(const v8h*)(rowp + base);
  v8h hi = *(const v8h*)(rowp + base + 16);
  return __builtin_shufflevector(lo, hi, 0,1,2,3,4,5,6,7,8,9,10,11,12,13,14,15);
}
__device__ __forceinline__ v8f wmma_f16(v16h a, v16h b, v8f c) {
  return __builtin_amdgcn_wmma_f32_16x16x32_f16(false, a, false, b, (short)0, c, false, false);
}
__device__ __forceinline__ float sigmoidf_(float x) { return 1.0f / (1.0f + expf(-x)); }

// LDS byte offset of a __shared__ object: generic ptr low 32 bits are the LDS
// address (flat-LDS aperture).
__device__ __forceinline__ unsigned lds_off(const void* p) {
  return (unsigned)(unsigned long long)p;
}
// CDNA5 async copy: 16 bytes global -> LDS, tracked by ASYNCcnt.
__device__ __forceinline__ void async_g2l_b128(unsigned lds_byte, const void* gsrc) {
  asm volatile("global_load_async_to_lds_b128 %0, %1, off"
               :: "v"(lds_byte), "v"((unsigned long long)gsrc) : "memory");
}

// ---------------------------------------------------------------------------
// Shared GEMM core: block = 8 waves (256 thr), each wave owns one 16-row tile,
// all waves share the same 4 column tiles.  B (packed WMMA layout) is staged
// into LDS with double-buffered async copies; 4 KB per K-step, loaded once
// per block instead of once per wave.
template<int KT>
__device__ __forceinline__ void gemm_core(const h16* __restrict__ rowp,
                                          const h16* __restrict__ Bp,
                                          const long* btile,      // 4 tile base offsets (halves)
                                          h16* sb,                // __shared__ h16[4096]
                                          v8f acc[4]) {
  const int tid  = threadIdx.x;
  const int lane = tid & 31;
  const int t    = tid >> 6;        // which of 4 tiles this thread stages
  const int c    = tid & 63;        // 16-byte chunk within the 512-half tile
  const h16* gsrc = Bp + btile[t] + (long)c * 8;
  const unsigned dbase = lds_off(sb) + (unsigned)(t * 1024 + c * 16); // bytes

  // prologue: stage kt=0 into buffer 0
  async_g2l_b128(dbase, gsrc);

  for (int kt = 0; kt < KT; ++kt) {
    __syncthreads();   // all reads of the buffer we are about to overwrite are done
    if (kt + 1 < KT) {
      async_g2l_b128(dbase + (((kt + 1) & 1) ? 4096u : 0u), gsrc + (long)(kt + 1) * 512);
      asm volatile("s_wait_asynccnt 0x1" ::: "memory"); // in-order: kt's stage done
    } else {
      asm volatile("s_wait_asynccnt 0x0" ::: "memory");
    }
    __syncthreads();   // every wave's slice of buffer[kt&1] is visible

    __builtin_prefetch(rowp + (kt + 2) * 32, 0, 0);
    v16h a = load_a(rowp, kt * 32, lane);
    const h16* bb = sb + ((kt & 1) ? 2048 : 0) + lane * 16;
    v16h b0 = *(const v16h*)(bb + 0 * 512);
    v16h b1 = *(const v16h*)(bb + 1 * 512);
    v16h b2 = *(const v16h*)(bb + 2 * 512);
    v16h b3 = *(const v16h*)(bb + 3 * 512);
    acc[0] = wmma_f16(a, b0, acc[0]);
    acc[1] = wmma_f16(a, b1, acc[1]);
    acc[2] = wmma_f16(a, b2, acc[2]);
    acc[3] = wmma_f16(a, b3, acc[3]);
  }
}

// ---------------------------------------------------------------------------
// fp32 -> f16 convert
__global__ void cvt_f16_kernel(const float* __restrict__ src, h16* __restrict__ dst, long n) {
  long i = (long)blockIdx.x * blockDim.x + threadIdx.x;
  if (i < n) dst[i] = (h16)src[i];
}

// Pack fp32 weight [K][N] into WMMA-B f16 tiles: out[((nt*KT+kt)*32+lane)*16+e]
__global__ void pack_b_kernel(const float* __restrict__ src, h16* __restrict__ dst,
                              int K, int Ncol) {
  long o = (long)blockIdx.x * blockDim.x + threadIdx.x;
  long tot = (long)K * Ncol;
  if (o >= tot) return;
  int e    = (int)(o & 15);
  int lane = (int)((o >> 4) & 31);
  long rest = o >> 9;
  int KT = K >> 5;
  int kt = (int)(rest % KT);
  int nt = (int)(rest / KT);
  int kk = kt * 32 + (e & 7) + ((e & 8) ? 16 : 0) + ((lane & 16) ? 8 : 0);
  int nn = nt * 16 + (lane & 15);
  dst[o] = (h16)src[(long)kk * Ncol + nn];
}

// ---------------------------------------------------------------------------
// GEMM1: Z[8192x1024] = x_f16[8192x1024] @ wword_pk    (K=1024)
__global__ void gemm1_kernel(const h16* __restrict__ A, const h16* __restrict__ Bp,
                             float* __restrict__ Z) {
  __shared__ h16 sb[4096];
  const int KT = 32;
  int lane = threadIdx.x & 31;
  int wave = threadIdx.x >> 5;
  int row0 = (blockIdx.x * 8 + wave) * 16;
  int n0   = blockIdx.y * 64;
  int nt0  = n0 >> 4;
  const h16* rowp = A + (long)(row0 + (lane & 15)) * 1024;
  long btile[4] = { (long)(nt0 + 0) * KT * 512, (long)(nt0 + 1) * KT * 512,
                    (long)(nt0 + 2) * KT * 512, (long)(nt0 + 3) * KT * 512 };
  v8f acc[4] = {{}, {}, {}, {}};
  gemm_core<KT>(rowp, Bp, btile, sb, acc);

  int col   = n0 + (lane & 15);
  int rbase = row0 + ((lane & 16) ? 8 : 0);
  for (int v = 0; v < 8; ++v) {
    float* zr = Z + (long)(rbase + v) * 1024 + col;
    zr[0] = acc[0][v]; zr[16] = acc[1][v]; zr[32] = acc[2][v]; zr[48] = acc[3][v];
  }
}

// ---------------------------------------------------------------------------
__device__ __forceinline__ float block_sum(float v, float* sdata) {
  int t = threadIdx.x;
  sdata[t] = v; __syncthreads();
  for (int off = 128; off > 0; off >>= 1) {
    if (t < off) sdata[t] += sdata[t + off];
    __syncthreads();
  }
  float r = sdata[0]; __syncthreads();
  return r;
}

// LN over Z rows (+b_word) -> H fp32 and Hh f16
__global__ void ln1_kernel(const float* __restrict__ Z, const float* __restrict__ bw,
                           const float* __restrict__ g, const float* __restrict__ bb,
                           float* __restrict__ H, h16* __restrict__ Hh) {
  __shared__ float sdata[256];
  long row = blockIdx.x;
  const float* zr = Z + row * 1024;
  float v[4]; float s = 0.f;
  for (int i = 0; i < 4; ++i) { int d = threadIdx.x * 4 + i; v[i] = zr[d] + bw[d]; s += v[i]; }
  float mu = block_sum(s, sdata) * (1.0f / 1024.0f);
  float q = 0.f;
  for (int i = 0; i < 4; ++i) { float c = v[i] - mu; q += c * c; }
  float var = block_sum(q, sdata) * (1.0f / 1024.0f);
  float inv = rsqrtf(var + 1e-5f);
  for (int i = 0; i < 4; ++i) {
    int d = threadIdx.x * 4 + i;
    float o = (v[i] - mu) * inv * g[d] + bb[d];
    H[row * 1024 + d]  = o;
    Hh[row * 1024 + d] = (h16)o;
  }
}

// ---------------------------------------------------------------------------
// GEMM2: inter[8176x4096] = gelu( concat(l,r)[8176x2048] @ w1_pk + b1 ), f16 out.
// A-row for logical row r=(n*511+s) is &Hh[(n*512+s)*1024], contiguous length 2048.
__global__ void gemm2_kernel(const h16* __restrict__ Hh, const h16* __restrict__ Bp,
                             const float* __restrict__ bias, h16* __restrict__ inter) {
  __shared__ h16 sb[4096];
  const int KT = 64;
  int lane = threadIdx.x & 31;
  int wave = threadIdx.x >> 5;
  int rt   = blockIdx.x * 8 + wave;
  bool valid = rt < 511;
  int rtc  = valid ? rt : 510;
  int row0 = rtc * 16;
  int rl   = row0 + (lane & 15);
  int nrow = rl / SS, srow = rl % SS;
  const h16* rowp = Hh + ((long)nrow * S0 + srow) * 1024;
  int n0  = blockIdx.y * 64;
  int nt0 = n0 >> 4;
  long btile[4] = { (long)(nt0 + 0) * KT * 512, (long)(nt0 + 1) * KT * 512,
                    (long)(nt0 + 2) * KT * 512, (long)(nt0 + 3) * KT * 512 };
  v8f acc[4] = {{}, {}, {}, {}};
  gemm_core<KT>(rowp, Bp, btile, sb, acc);

  if (!valid) return;
  int rbase = row0 + ((lane & 16) ? 8 : 0);
  for (int t = 0; t < 4; ++t) {
    int col = n0 + t * 16 + (lane & 15);
    float bv = bias[col];
    for (int v = 0; v < 8; ++v) {
      float a  = acc[t][v] + bv;
      float ge = 0.5f * a * (1.0f + erff(a * 0.70710678118654752f));
      inter[(long)(rbase + v) * 4096 + col] = (h16)ge;
    }
  }
}

// ---------------------------------------------------------------------------
// GEMM3: contents = inter @ w2_pk + b2, fused sigmoid-gate combine -> U[8176x1024].
// Wave owns column tiles {d, d+1024, d+2048, d+3072}.
__global__ void gemm3_kernel(const h16* __restrict__ inter, const h16* __restrict__ Bp,
                             const float* __restrict__ bias, const float* __restrict__ H,
                             float* __restrict__ U) {
  __shared__ h16 sb[4096];
  const int KT = 128;
  int lane = threadIdx.x & 31;
  int wave = threadIdx.x >> 5;
  int rt   = blockIdx.x * 8 + wave;
  bool valid = rt < 511;
  int rtc  = valid ? rt : 510;
  int row0 = rtc * 16;
  const h16* rowp = inter + (long)(row0 + (lane & 15)) * 4096;
  int nt = blockIdx.y;                 // tile within the 1024-wide channel group
  long btile[4] = { (long)(nt +   0) * KT * 512, (long)(nt +  64) * KT * 512,
                    (long)(nt + 128) * KT * 512, (long)(nt + 192) * KT * 512 };
  v8f acc[4] = {{}, {}, {}, {}};
  gemm_core<KT>(rowp, Bp, btile, sb, acc);

  if (!valid) return;
  int d     = nt * 16 + (lane & 15);
  int rbase = row0 + ((lane & 16) ? 8 : 0);
  float bd0 = bias[d], bd1 = bias[d + 1024], bd2 = bias[d + 2048], bd3 = bias[d + 3072];
  for (int v = 0; v < 8; ++v) {
    int r = rbase + v;
    int n = r / SS, s = r % SS;
    float f1 = sigmoidf_(acc[0][v] + bd0);
    float f2 = sigmoidf_(acc[1][v] + bd1);
    float ig = sigmoidf_(acc[2][v] + bd2);
    float pa = acc[3][v] + bd3;
    float lv = H[((long)n * S0 + s) * 1024 + d];
    float rv = H[((long)n * S0 + s + 1) * 1024 + d];
    U[(long)r * 1024 + d] = f1 * lv + f2 * rv + ig * pa;
  }
}

// ---------------------------------------------------------------------------
// LN2 over U rows -> NH fp32; fused decoder dot -> comp[r]
__global__ void ln2_kernel(const float* __restrict__ U, const float* __restrict__ g,
                           const float* __restrict__ bb, const float* __restrict__ wdec,
                           const float* __restrict__ bdec, float* __restrict__ NH,
                           float* __restrict__ comp) {
  __shared__ float sdata[256];
  long row = blockIdx.x;
  const float* ur = U + row * 1024;
  float v[4]; float s = 0.f;
  for (int i = 0; i < 4; ++i) { int d = threadIdx.x * 4 + i; v[i] = ur[d]; s += v[i]; }
  float mu = block_sum(s, sdata) * (1.0f / 1024.0f);
  float q = 0.f;
  for (int i = 0; i < 4; ++i) { float c = v[i] - mu; q += c * c; }
  float var = block_sum(q, sdata) * (1.0f / 1024.0f);
  float inv = rsqrtf(var + 1e-5f);
  float p = 0.f;
  for (int i = 0; i < 4; ++i) {
    int d = threadIdx.x * 4 + i;
    float o = (v[i] - mu) * inv * g[d] + bb[d];
    NH[row * 1024 + d] = o;
    p += o * wdec[d];
  }
  float tot = block_sum(p, sdata);
  if (threadIdx.x == 0) comp[row] = tot + bdec[0];
}

// ---------------------------------------------------------------------------
// Top-5 argmax per batch row (softmax is monotone -> rank comp directly).
// Tie-break: lower index wins (matches jax.lax.top_k).
__global__ void top5_kernel(const float* __restrict__ comp, int* __restrict__ idxout) {
  __shared__ float sv[512];
  __shared__ int   si[512];
  int n = blockIdx.x;
  int t = threadIdx.x;
  float v = (t < SS) ? comp[n * SS + t] : -INFINITY;
  for (int k = 0; k < 5; ++k) {
    sv[t] = v; si[t] = t;
    __syncthreads();
    for (int off = 256; off > 0; off >>= 1) {
      if (t < off) {
        float v2 = sv[t + off]; int i2 = si[t + off];
        if (v2 > sv[t] || (v2 == sv[t] && i2 < si[t])) { sv[t] = v2; si[t] = i2; }
      }
      __syncthreads();
    }
    int sel = si[0];
    if (t == 0) idxout[n * 5 + k] = sel;
    if (t == sel) v = -INFINITY;
    __syncthreads();
  }
}

// ---------------------------------------------------------------------------
// Assemble h_next[n,k,s,:] : new_h at the selected slot, h[s] left of it, h[s+1] right.
__global__ void assemble_kernel(const float* __restrict__ H, const float* __restrict__ NH,
                                const int* __restrict__ idx, float* __restrict__ out) {
  int s = blockIdx.x, k = blockIdx.y, n = blockIdx.z;
  int sel = idx[n * 5 + k];
  const float* src;
  if (s == sel)     src = NH + ((long)n * SS + s) * 1024;
  else if (s < sel) src = H + ((long)n * S0 + s) * 1024;
  else              src = H + ((long)n * S0 + s + 1) * 1024;
  float4*       dp = (float4*)(out + (((long)(n * 5 + k) * SS) + s) * 1024);
  const float4* sp = (const float4*)src;
  dp[threadIdx.x] = sp[threadIdx.x];
}

// ---------------------------------------------------------------------------
extern "C" void kernel_launch(void* const* d_in, const int* in_sizes, int n_in,
                              void* d_out, int out_size, void* d_ws, size_t ws_size,
                              hipStream_t stream) {
  (void)in_sizes; (void)n_in; (void)out_size; (void)ws_size;
  const float* x     = (const float*)d_in[0];
  const float* wword = (const float*)d_in[1];
  const float* bword = (const float*)d_in[2];
  const float* lng   = (const float*)d_in[3];
  const float* lnb   = (const float*)d_in[4];
  const float* w1    = (const float*)d_in[5];
  const float* b1    = (const float*)d_in[6];
  const float* w2    = (const float*)d_in[7];
  const float* b2    = (const float*)d_in[8];
  const float* ln2g  = (const float*)d_in[9];
  const float* ln2b  = (const float*)d_in[10];
  const float* wdec  = (const float*)d_in[11];
  const float* bdec  = (const float*)d_in[12];
  float* out = (float*)d_out;

  char* wsp = (char*)d_ws;
  auto carve = [&](size_t bytes) -> char* {
    char* p = wsp;
    wsp += (bytes + 255) & ~(size_t)255;
    return p;
  };
  const long MX = (long)NB * S0;        // 8192 rows
  h16*   x_h   = (h16*)  carve(MX * 1024 * sizeof(h16));
  h16*   ww_pk = (h16*)  carve((long)1024 * 1024 * sizeof(h16));
  h16*   w1_pk = (h16*)  carve((long)2048 * 4096 * sizeof(h16));
  h16*   w2_pk = (h16*)  carve((long)4096 * 4096 * sizeof(h16));
  float* Z     = (float*)carve(MX * 1024 * sizeof(float));
  float* H     = (float*)carve(MX * 1024 * sizeof(float));
  h16*   Hh    = (h16*)  carve(MX * 1024 * sizeof(h16));
  h16*   inter = (h16*)  carve((long)MROW2 * 4096 * sizeof(h16));
  float* U     = (float*)carve((long)MROW2 * 1024 * sizeof(float));
  float* NH    = (float*)carve((long)MROW2 * 1024 * sizeof(float));
  float* comp  = (float*)carve((long)MROW2 * sizeof(float));
  int*   idxb  = (int*)  carve(NB * 5 * sizeof(int));

  // 0) precision conversion + WMMA-B packing
  {
    long n = MX * 1024;
    cvt_f16_kernel<<<dim3((n + 255) / 256), dim3(256), 0, stream>>>(x, x_h, n);
  }
  pack_b_kernel<<<dim3((1024L * 1024 + 255) / 256), dim3(256), 0, stream>>>(wword, ww_pk, 1024, 1024);
  pack_b_kernel<<<dim3((2048L * 4096 + 255) / 256), dim3(256), 0, stream>>>(w1, w1_pk, 2048, 4096);
  pack_b_kernel<<<dim3((4096L * 4096 + 255) / 256), dim3(256), 0, stream>>>(w2, w2_pk, 4096, 4096);

  // 1) word projection GEMM + LN
  gemm1_kernel<<<dim3(64, 16), dim3(256), 0, stream>>>(x_h, ww_pk, Z);
  ln1_kernel<<<dim3((unsigned)MX), dim3(256), 0, stream>>>(Z, bword, lng, lnb, H, Hh);

  // 2) [l,r] @ w1 + gelu (concat is free: rows s,s+1 are contiguous)
  gemm2_kernel<<<dim3(64, 64), dim3(256), 0, stream>>>(Hh, w1_pk, b1, inter);

  // 3) inter @ w2 with fused sigmoid-gated combine
  gemm3_kernel<<<dim3(64, 64), dim3(256), 0, stream>>>(inter, w2_pk, b2, H, U);

  // 4) LN2 + decoder dot
  ln2_kernel<<<dim3(MROW2), dim3(256), 0, stream>>>(U, ln2g, ln2b, wdec, bdec, NH, comp);

  // 5) top-5 per batch row (softmax monotone -> skip it)
  top5_kernel<<<dim3(NB), dim3(512), 0, stream>>>(comp, idxb);

  // 6) gather/assemble the beam outputs
  assemble_kernel<<<dim3(SS, 5, NB), dim3(256), 0, stream>>>(H, NH, idxb, out);
}
